// SpatialPoolingLayer_74921409511464
// MI455X (gfx1250) — 8-bit matrix operand order repaired and checked by reading, compiled, not run on hardware
//
#include <hip/hip_runtime.h>

#define IN_DIM   8192
#define OUT_DIM  16384
#define TOPK_K   327
#define COLS_PB  64
#define ROWSPLIT 8
#define ROWS_PB  (IN_DIM / ROWSPLIT)
#define CHUNK    64

typedef int   v8i __attribute__((ext_vector_type(8)));
typedef int   i4v __attribute__((ext_vector_type(4)));
typedef float f4v __attribute__((ext_vector_type(4)));

__global__ __launch_bounds__(256) void sp_zero(int* __restrict__ overlap) {
    int i = blockIdx.x * blockDim.x + threadIdx.x;
    if (i < OUT_DIM) overlap[i] = 0;
}

__global__ __launch_bounds__(128) void sp_overlap(const int* __restrict__ x,
                                                  const float* __restrict__ p,
                                                  int* __restrict__ overlap) {
    __shared__ __align__(16) unsigned char bits[COLS_PB * CHUNK];
    const int t    = threadIdx.x;
    const int lane = t & 31;
    const int wave = t >> 5;
    const int n    = lane & 15;
    const int h    = lane >> 4;
    const int j0   = blockIdx.x * COLS_PB;
    const int ib   = blockIdx.y * ROWS_PB;

    v8i acc = {};
    const f4v fzero = {0.0f, 0.0f, 0.0f, 0.0f};

    for (int c0 = 0; c0 < ROWS_PB; c0 += CHUNK) {
        const int i0 = ib + c0;
        #pragma unroll
        for (int pass = 0; pass < 2; ++pass) {
            const int q   = t + pass * 128;
            const int rg  = q >> 4;
            const int cq  = q & 15;
            const int row = rg * 4;
            const i4v xv  = *(const i4v*)(x + i0 + row);
            f4v pv[4];
            #pragma unroll
            for (int rr = 0; rr < 4; ++rr) {
                if (xv[rr] != 0) {
                    pv[rr] = __builtin_nontemporal_load(
                        (const f4v*)(p + (size_t)(i0 + row + rr) * OUT_DIM + (j0 + cq * 4)));
                } else {
                    pv[rr] = fzero;
                }
            }
            if (c0 + CHUNK < ROWS_PB) {
                const i4v xn = *(const i4v*)(x + i0 + CHUNK + row);
                #pragma unroll
                for (int rr = 0; rr < 4; ++rr) {
                    if (xn[rr] != 0) {
                        __builtin_prefetch(
                            p + (size_t)(i0 + CHUNK + row + rr) * OUT_DIM + (j0 + cq * 4), 0, 1);
                    }
                }
            }
            #pragma unroll
            for (int cc = 0; cc < 4; ++cc) {
                unsigned int d = 0;
                #pragma unroll
                for (int rr = 0; rr < 4; ++rr) {
                    d |= (pv[rr][cc] > 0.5f ? 1u : 0u) << (8 * rr);
                }
                *(unsigned int*)&bits[(cq * 4 + cc) * CHUNK + row] = d;
            }
        }
        __syncthreads();

        v8i a;
        #pragma unroll
        for (int v = 0; v < 8; ++v) {
            const int base = ((v >> 2) << 5) + (h << 4) + ((v & 3) << 2);
            const i4v xv = *(const i4v*)(x + i0 + base);
            unsigned int d = (unsigned)(xv[0] != 0)
                           | ((unsigned)(xv[1] != 0) << 8)
                           | ((unsigned)(xv[2] != 0) << 16)
                           | ((unsigned)(xv[3] != 0) << 24);
            a[v] = (int)d;
        }
        const unsigned char* colp = &bits[(wave * 16 + n) * CHUNK];
        const i4v blo = *(const i4v*)(colp + h * 16);
        const i4v bhi = *(const i4v*)(colp + 32 + h * 16);
        v8i b;
        b[0] = blo[0]; b[1] = blo[1]; b[2] = blo[2]; b[3] = blo[3];
        b[4] = bhi[0]; b[5] = bhi[1]; b[6] = bhi[2]; b[7] = bhi[3];

        acc = __builtin_amdgcn_wmma_i32_16x16x64_iu8(
             false, a,  false, b, acc,
             false,  false);

        __syncthreads();
    }

    if (lane < 16) atomicAdd(&overlap[j0 + wave * 16 + n], acc[0]);
}

__global__ __launch_bounds__(1024) void sp_topk(const int* __restrict__ overlap,
                                                float* __restrict__ out) {
    __shared__ int hist[IN_DIM + 1];
    __shared__ int scn[1024];
    __shared__ int sT;
    __shared__ int sNeed;
    const int t = threadIdx.x;

    for (int i = t; i <= IN_DIM; i += 1024) hist[i] = 0;
    __syncthreads();

    int v[16];
    #pragma unroll
    for (int u = 0; u < 16; ++u) {
        v[u] = overlap[t * 16 + u];
        atomicAdd(&hist[v[u]], 1);
    }
    __syncthreads();

    int cs = 0;
    #pragma unroll
    for (int b = 0; b < 8; ++b) cs += hist[t * 8 + b];
    if (t == 1023) cs += hist[IN_DIM];
    scn[t] = cs;
    __syncthreads();
    for (int off = 1; off < 1024; off <<= 1) {
        const int add = (t + off < 1024) ? scn[t + off] : 0;
        __syncthreads();
        scn[t] += add;
        __syncthreads();
    }

    {
        const int Sc = scn[t];
        const int Sn = (t < 1023) ? scn[t + 1] : hist[IN_DIM];
        if (Sc >= TOPK_K && Sn < TOPK_K) {
            int s = (t < 1023) ? scn[t + 1] : 0;
            const int vhi = (t == 1023) ? IN_DIM : t * 8 + 7;
            for (int vv = vhi; vv >= t * 8; --vv) {
                s += hist[vv];
                if (s >= TOPK_K) {
                    sT    = vv;
                    sNeed = TOPK_K - (s - hist[vv]);
                    break;
                }
            }
        }
    }
    __syncthreads();
    const int T = sT, Need = sNeed;

    int tc = 0;
    #pragma unroll
    for (int u = 0; u < 16; ++u) tc += (v[u] == T) ? 1 : 0;
    scn[t] = tc;
    __syncthreads();
    for (int off = 1; off < 1024; off <<= 1) {
        const int add = (t >= off) ? scn[t - off] : 0;
        __syncthreads();
        scn[t] += add;
        __syncthreads();
    }
    int budget = Need - (scn[t] - tc);

    #pragma unroll
    for (int u = 0; u < 16; ++u) {
        float o = 0.0f;
        if (v[u] > T) {
            o = 1.0f;
        } else if (v[u] == T && budget > 0) {
            o = 1.0f;
            --budget;
        }
        out[t * 16 + u] = o;
    }
}

extern "C" void kernel_launch(void* const* d_in, const int* in_sizes, int n_in,
                              void* d_out, int out_size, void* d_ws, size_t ws_size,
                              hipStream_t stream) {
    (void)in_sizes; (void)n_in; (void)out_size; (void)ws_size;
    const int*   x = (const int*)d_in[0];
    const float* p = (const float*)d_in[1];
    float* out     = (float*)d_out;
    int* overlap   = (int*)d_ws;

    sp_zero<<<(OUT_DIM + 255) / 256, 256, 0, stream>>>(overlap);
    sp_overlap<<<dim3(OUT_DIM / COLS_PB, ROWSPLIT), 128, 0, stream>>>(x, p, overlap);
    sp_topk<<<1, 1024, 0, stream>>>(overlap, out);
}
